// Baller2VecSeq2Seq_22058952032835
// MI455X (gfx1250) — matrix operand order repaired and checked by reading, not compiled, not run on hardware
//
#include <hip/hip_runtime.h>
#include <hip/hip_bf16.h>
#include <math.h>

// ---------------------------------------------------------------------------
// Baller2Vec seq2seq transformer for gfx1250 (MI455X), WMMA f16 path.
//
// All matmuls run through v_wmma_f32_16x16x32_f16 (f16 in, f32 accumulate).
// B operands are always [N][K] row-major: weights converted transposed once at
// launch; V written transposed by its projection GEMM. GEMM uses double-
// buffered LDS with a software pipeline (stage tile t+1 during tile-t WMMAs)
// and uniform interior/edge load paths (no per-chunk exec masking interior).
//
// Input flattening assumption: setup_inputs() dict order (Python insertion
// order), nested dicts in insertion order, lists by index.
// ---------------------------------------------------------------------------

typedef __attribute__((ext_vector_type(16))) _Float16 v16h;
typedef __attribute__((ext_vector_type(8)))  float    v8f;
typedef __attribute__((ext_vector_type(4)))  unsigned int u32x4;

#define SEQ_LEN   512
#define EMBED     32
#define DMODEL    512
#define NHEAD     8
#define HD        64
#define DFF       2048
#define NLAYERS   4
#define MLP_IN    35
#define MLP_HID   256
#define NLAB      121
#define NTOK      3072
#define NPL_TOK   2560
#define FEAT_LD   40
#define SQRT_D    22.62741699796952f

#define BM 128
#define BN 128
#define BK 32
#define LDA_S (BK + 8)   // LDS row stride for A tile (halves)
#define LDB_S (BK + 8)   // LDS row stride for B tile (halves), [n][k] layout

// ---------------------------------------------------------------------------
__device__ __forceinline__ int tok_step(int i) {
    return (i < NPL_TOK) ? (i / 5) : (i - NPL_TOK);
}

// A fragment (16x32 f16, MxK) from LDS row-major [m][k], stride ldk halves.
// ISA 7.12.2: lanes 0-15 -> M=0..15, halves {0..7}=K0..7, {8..15}=K16..23;
//             lanes 16-31 -> halves {0..7}=K8..15, {8..15}=K24..31.
__device__ __forceinline__ v16h frag_a(const _Float16* s, int ldk) {
    const int lane = threadIdx.x & 31;
    const int m  = lane & 15;
    const int kb = (lane >> 4) * 8;
    const _Float16* p = s + m * ldk + kb;
    v16h a;
#pragma unroll
    for (int h = 0; h < 8; ++h) { a[h] = p[h]; a[h + 8] = p[h + 16]; }
    return a;
}

// B fragment (32x16 f16, KxN) from LDS stored transposed [n][k], stride ldk.
// lanes 0-15 -> N=0..15 hold K=0..7,16..23 ; lanes 16-31 hold K=8..15,24..31.
__device__ __forceinline__ v16h frag_b(const _Float16* s, int ldk) {
    const int lane = threadIdx.x & 31;
    const int n  = lane & 15;
    const int kb = (lane >> 4) * 8;
    const _Float16* p = s + n * ldk + kb;
    v16h b;
#pragma unroll
    for (int h = 0; h < 8; ++h) { b[h] = p[h]; b[h + 8] = p[h + 16]; }
    return b;
}

// ---------------------------------------------------------------------------
// WMMA GEMM: C = act(alpha * (A[MxK] @ B^T + bias)), B stored [N][K] row-major.
// Apad/Bpad: valid (zero-padded) halves from each row start.
// store_ct: write the f16 output transposed (C16[col*ldc16 + row]).
// Double-buffered LDS + software pipeline; one barrier per k-tile.
// ---------------------------------------------------------------------------
__global__ __launch_bounds__(256)
void b2v_gemm_f16(const _Float16* __restrict__ A, int lda, int Apad,
                  const _Float16* __restrict__ B, int ldb, int Bpad,
                  float* __restrict__ C, int ldc,
                  _Float16* __restrict__ C16, int ldc16, int store_ct,
                  const float* __restrict__ bias,
                  int M, int N, int K, int relu, float alpha)
{
    __shared__ __align__(16) _Float16 sA[2][BM][LDA_S];
    __shared__ __align__(16) _Float16 sB[2][BN][LDB_S];   // [n][k]

    const int tid  = threadIdx.x;
    const int lane = tid & 31;
    const int wave = tid >> 5;
    const int wm   = wave >> 2;   // 0..1  (64 rows each)
    const int wn   = wave & 3;    // 0..3  (32 cols each)
    const int m0   = blockIdx.y * BM;
    const int n0   = blockIdx.x * BN;

    // per-thread chunk coordinates (2 chunks of 8 halves for A, 2 for B)
    const int ar0 = tid >> 2,        ak0 = (tid & 3) * 8;
    const int ar1 = (tid + 256) >> 2, ak1 = ak0;          // same koff pattern
    const _Float16* pA0 = A + (size_t)(m0 + ar0) * lda + ak0;
    const _Float16* pA1 = A + (size_t)(m0 + ar1) * lda + ak1;
    const _Float16* pB0 = B + (size_t)(n0 + ar0) * ldb + ak0;
    const _Float16* pB1 = B + (size_t)(n0 + ar1) * ldb + ak1;

    const int  ntiles = (K + BK - 1) / BK;
    const bool intA   = (m0 + BM <= M) && (ntiles * BK <= Apad);
    const bool intB   = (n0 + BN <= N) && (ntiles * BK <= Bpad);

    const u32x4 zv = {0u, 0u, 0u, 0u};
    u32x4 ra0, ra1, rb0, rb1;

    // ---- stage tile 0 into registers ----
    if (intA) {
        ra0 = *(const u32x4*)(pA0);
        ra1 = *(const u32x4*)(pA1);
    } else {
        ra0 = zv; ra1 = zv;
        if (m0 + ar0 < M && ak0 + 8 <= Apad) ra0 = *(const u32x4*)(pA0);
        if (m0 + ar1 < M && ak1 + 8 <= Apad) ra1 = *(const u32x4*)(pA1);
    }
    if (intB) {
        rb0 = *(const u32x4*)(pB0);
        rb1 = *(const u32x4*)(pB1);
    } else {
        rb0 = zv; rb1 = zv;
        if (n0 + ar0 < N && ak0 + 8 <= Bpad) rb0 = *(const u32x4*)(pB0);
        if (n0 + ar1 < N && ak1 + 8 <= Bpad) rb1 = *(const u32x4*)(pB1);
    }
    *(u32x4*)(&sA[0][ar0][ak0]) = ra0;
    *(u32x4*)(&sA[0][ar1][ak1]) = ra1;
    *(u32x4*)(&sB[0][ar0][ak0]) = rb0;
    *(u32x4*)(&sB[0][ar1][ak1]) = rb1;

    v8f zero;
#pragma unroll
    for (int r = 0; r < 8; ++r) zero[r] = 0.0f;
    v8f acc[4][2];
#pragma unroll
    for (int i = 0; i < 4; ++i)
#pragma unroll
        for (int j = 0; j < 2; ++j) acc[i][j] = zero;

    for (int t = 0; t < ntiles; ++t) {
        __syncthreads();               // buf[t&1] tile visible to all waves

        const int  k1   = (t + 1) * BK;
        const bool more = k1 < K;
        if (more) {                    // stage next tile while computing
            if (intA) {
                ra0 = *(const u32x4*)(pA0 + k1);
                ra1 = *(const u32x4*)(pA1 + k1);
            } else {
                ra0 = zv; ra1 = zv;
                if (m0 + ar0 < M && k1 + ak0 + 8 <= Apad) ra0 = *(const u32x4*)(pA0 + k1);
                if (m0 + ar1 < M && k1 + ak1 + 8 <= Apad) ra1 = *(const u32x4*)(pA1 + k1);
            }
            if (intB) {
                rb0 = *(const u32x4*)(pB0 + k1);
                rb1 = *(const u32x4*)(pB1 + k1);
            } else {
                rb0 = zv; rb1 = zv;
                if (n0 + ar0 < N && k1 + ak0 + 8 <= Bpad) rb0 = *(const u32x4*)(pB0 + k1);
                if (n0 + ar1 < N && k1 + ak1 + 8 <= Bpad) rb1 = *(const u32x4*)(pB1 + k1);
            }
        }

        const int buf = t & 1;
        v16h af[4];
#pragma unroll
        for (int ti = 0; ti < 4; ++ti)
            af[ti] = frag_a(&sA[buf][wm * 64 + ti * 16][0], LDA_S);
        v16h bf[2];
#pragma unroll
        for (int tj = 0; tj < 2; ++tj)
            bf[tj] = frag_b(&sB[buf][wn * 32 + tj * 16][0], LDB_S);

#pragma unroll
        for (int ti = 0; ti < 4; ++ti)
#pragma unroll
            for (int tj = 0; tj < 2; ++tj)
                acc[ti][tj] = __builtin_amdgcn_wmma_f32_16x16x32_f16(
                    false, af[ti], false, bf[tj], (short)0, acc[ti][tj],
                    false, false);

        if (more) {                    // fill the other buffer
            const int nb = buf ^ 1;
            *(u32x4*)(&sA[nb][ar0][ak0]) = ra0;
            *(u32x4*)(&sA[nb][ar1][ak1]) = ra1;
            *(u32x4*)(&sB[nb][ar0][ak0]) = rb0;
            *(u32x4*)(&sB[nb][ar1][ak1]) = rb1;
        }
    }

    // epilogue: C/D layout — VGPR r holds M = r + 8*(lane>>4), N = lane&15
    const int rsel = (lane >> 4) * 8;
    const int ncol = lane & 15;
#pragma unroll
    for (int ti = 0; ti < 4; ++ti)
#pragma unroll
        for (int tj = 0; tj < 2; ++tj)
#pragma unroll
            for (int r = 0; r < 8; ++r) {
                const int row = m0 + wm * 64 + ti * 16 + rsel + r;
                const int col = n0 + wn * 32 + tj * 16 + ncol;
                if (row < M && col < N) {
                    float v = acc[ti][tj][r];
                    if (bias) v += bias[col];
                    v *= alpha;
                    if (relu) v = fmaxf(v, 0.0f);
                    if (C) C[(size_t)row * ldc + col] = v;
                    if (C16) {
                        if (store_ct) C16[(size_t)col * ldc16 + row] = (_Float16)v;
                        else          C16[(size_t)row * ldc16 + col] = (_Float16)v;
                    }
                }
            }
}

// ---------------------------------------------------------------------------
// Weight convert fp32 [K][N] -> f16 transposed [N][ldt], zero pad k in [K,ldt)
// One output row per blockIdx.y (no integer div in the loop).
// ---------------------------------------------------------------------------
__global__ void b2v_w_to_f16_t(const float* __restrict__ src,
                               _Float16* __restrict__ dst,
                               int K, int N, int ldt)
{
    const int n = blockIdx.y;
    for (int kp = blockIdx.x * blockDim.x + threadIdx.x; kp < ldt;
         kp += gridDim.x * blockDim.x) {
        dst[(size_t)n * ldt + kp] =
            (kp < K) ? (_Float16)src[(size_t)kp * N + n] : (_Float16)0.0f;
    }
}

// ---------------------------------------------------------------------------
// Feature assembly: [NTOK x FEAT_LD] f16 (35 real cols, zero padded to 40)
// ---------------------------------------------------------------------------
__global__ void b2v_feats(const float* __restrict__ pemb,
                          const int* __restrict__ pidx,
                          const float* __restrict__ pxs,
                          const float* __restrict__ pys,
                          const float* __restrict__ phs,
                          const float* __restrict__ bemb,
                          const float* __restrict__ bxs,
                          const float* __restrict__ bys,
                          const float* __restrict__ bzs,
                          int start, _Float16* __restrict__ out)
{
    const int tok = blockIdx.x;
    const int j   = threadIdx.x;
    if (j >= FEAT_LD) return;
    float v = 0.0f;
    if (tok < NPL_TOK) {
        const int t = tok / 5;
        const int p = start + (tok % 5);
        if (j < EMBED) {
            const int id = pidx[t * 10 + p];
            v = pemb[id * EMBED + j];
        } else if (j == EMBED)     v = pxs[t * 10 + p];
        else if (j == EMBED + 1)   v = pys[t * 10 + p];
        else if (j == EMBED + 2)   v = phs[t * 10 + p];
    } else {
        const int t = tok - NPL_TOK;
        if (j < EMBED)             v = bemb[j];
        else if (j == EMBED)       v = bxs[t];
        else if (j == EMBED + 1)   v = bys[t];
        else if (j == EMBED + 2)   v = bzs[t];
    }
    out[tok * FEAT_LD + j] = (_Float16)v;
}

// ---------------------------------------------------------------------------
// Row softmax with optional step-mask; fp32 in, f16 probabilities out.
// ---------------------------------------------------------------------------
__global__ __launch_bounds__(256)
void b2v_softmax(const float* __restrict__ S, _Float16* __restrict__ P,
                 int ncols, float scale, int masked)
{
    __shared__ float red[256];
    const int row = blockIdx.x;
    const int tid = threadIdx.x;
    const int rstep = masked ? tok_step(row) : 0;
    const float* srow = S + (size_t)row * ncols;

    float mx = -3.0e38f;
    for (int c = tid; c < ncols; c += 256) {
        const bool ok = !masked || (tok_step(c) <= rstep);
        if (ok) mx = fmaxf(mx, srow[c] * scale);
    }
    red[tid] = mx; __syncthreads();
    for (int off = 128; off > 0; off >>= 1) {
        if (tid < off) red[tid] = fmaxf(red[tid], red[tid + off]);
        __syncthreads();
    }
    mx = red[0]; __syncthreads();

    float sum = 0.0f;
    for (int c = tid; c < ncols; c += 256) {
        const bool ok = !masked || (tok_step(c) <= rstep);
        if (ok) sum += __expf(srow[c] * scale - mx);
    }
    red[tid] = sum; __syncthreads();
    for (int off = 128; off > 0; off >>= 1) {
        if (tid < off) red[tid] += red[tid + off];
        __syncthreads();
    }
    sum = red[0]; __syncthreads();
    const float inv = 1.0f / sum;

    _Float16* prow = P + (size_t)row * ncols;
    for (int c = tid; c < ncols; c += 256) {
        const bool ok = !masked || (tok_step(c) <= rstep);
        const float p = ok ? __expf(srow[c] * scale - mx) * inv : 0.0f;
        prow[c] = (_Float16)p;
    }
}

// ---------------------------------------------------------------------------
// Fused residual add + LayerNorm over DMODEL=512 cols; writes fp32 + f16.
// ---------------------------------------------------------------------------
__global__ __launch_bounds__(256)
void b2v_add_ln(const float* __restrict__ x, const float* __restrict__ dx,
                const float* __restrict__ g, const float* __restrict__ b,
                float* __restrict__ out, _Float16* __restrict__ out16)
{
    __shared__ float red[256];
    const int row = blockIdx.x;
    const int tid = threadIdx.x;
    const float* xr = x  + (size_t)row * DMODEL;
    const float* dr = dx + (size_t)row * DMODEL;

    const float v0 = xr[tid]       + dr[tid];
    const float v1 = xr[tid + 256] + dr[tid + 256];

    red[tid] = v0 + v1; __syncthreads();
    for (int off = 128; off > 0; off >>= 1) {
        if (tid < off) red[tid] += red[tid + off];
        __syncthreads();
    }
    const float mean = red[0] * (1.0f / DMODEL);
    __syncthreads();

    const float d0 = v0 - mean, d1 = v1 - mean;
    red[tid] = d0 * d0 + d1 * d1; __syncthreads();
    for (int off = 128; off > 0; off >>= 1) {
        if (tid < off) red[tid] += red[tid + off];
        __syncthreads();
    }
    const float var = red[0] * (1.0f / DMODEL);
    const float inv = rsqrtf(var + 1e-5f);

    const float o0 = g[tid]       * d0 * inv + b[tid];
    const float o1 = g[tid + 256] * d1 * inv + b[tid + 256];
    float* orow = out + (size_t)row * DMODEL;
    _Float16* orow16 = out16 + (size_t)row * DMODEL;
    orow[tid]       = o0;  orow16[tid]       = (_Float16)o0;
    orow[tid + 256] = o1;  orow16[tid + 256] = (_Float16)o1;
}

// ---------------------------------------------------------------------------
// Host side
// ---------------------------------------------------------------------------
struct AttnW { const _Float16 *wq, *wk, *wv, *wo; const float *bq, *bk, *bv, *bo; };
struct LinW  { const _Float16* w; const float* b; int ldt; };
struct LNW   { const float *g, *b; };
struct EncL  { AttnW sa; LNW ln1, ln2; LinW ff1, ff2; };
struct DecL  { AttnW sa, ca; LNW ln1, ln2, ln3; LinW ff1, ff2; };

static void gemm(hipStream_t s, const _Float16* A, int lda, int Apad,
                 const _Float16* B, int ldb, int Bpad,
                 float* C, int ldc, _Float16* C16, int ldc16, int store_ct,
                 const float* bias, int M, int N, int K, int relu, float alpha)
{
    dim3 grid((N + BN - 1) / BN, (M + BM - 1) / BM);
    b2v_gemm_f16<<<grid, 256, 0, s>>>(A, lda, Apad, B, ldb, Bpad,
                                      C, ldc, C16, ldc16, store_ct, bias,
                                      M, N, K, relu, alpha);
}

static void run_mha(hipStream_t s, const _Float16* q_in, const _Float16* kv_in,
                    int Sq, int Sk, const AttnW& w,
                    float* Sbuf, _Float16* Pbuf,
                    _Float16* Q, _Float16* Kb, _Float16* VT, _Float16* O,
                    float* delta, int masked)
{
    // projections; V written transposed [DMODEL][Sk]
    gemm(s, q_in,  DMODEL, DMODEL, w.wq, DMODEL, DMODEL, nullptr, 0,
         Q,  DMODEL, 0, w.bq, Sq, DMODEL, DMODEL, 0, 1.0f);
    gemm(s, kv_in, DMODEL, DMODEL, w.wk, DMODEL, DMODEL, nullptr, 0,
         Kb, DMODEL, 0, w.bk, Sk, DMODEL, DMODEL, 0, 1.0f);
    gemm(s, kv_in, DMODEL, DMODEL, w.wv, DMODEL, DMODEL, nullptr, 0,
         VT, NTOK, 1, w.bv, Sk, DMODEL, DMODEL, 0, 1.0f);
    for (int h = 0; h < NHEAD; ++h) {
        // scores = Q_h @ K_h^T ; both [S][HD] slices (contiguous along HD)
        gemm(s, Q + h * HD, DMODEL, HD, Kb + h * HD, DMODEL, HD,
             Sbuf, Sk, nullptr, 0, 0, nullptr, Sq, Sk, HD, 0, 1.0f);
        b2v_softmax<<<Sq, 256, 0, s>>>(Sbuf, Pbuf, Sk, 0.125f, masked);
        // O_h = P @ V_h ; V^T rows h*HD.. are [HD][Sk] (contiguous along Sk)
        gemm(s, Pbuf, Sk, Sk, VT + (size_t)h * HD * NTOK, NTOK, Sk,
             nullptr, 0, O + h * HD, DMODEL, 0, nullptr, Sq, HD, Sk, 0, 1.0f);
    }
    gemm(s, O, DMODEL, DMODEL, w.wo, DMODEL, DMODEL, delta, DMODEL,
         nullptr, 0, 0, w.bo, Sq, DMODEL, DMODEL, 0, 1.0f);
}

extern "C" void kernel_launch(void* const* d_in, const int* in_sizes, int n_in,
                              void* d_out, int out_size, void* d_ws, size_t ws_size,
                              hipStream_t stream)
{
    (void)in_sizes; (void)n_in; (void)out_size; (void)ws_size;

    // ---- workspace bump allocator -------------------------------------
    char* wsp = (char*)d_ws;
    auto alloc = [&](size_t bytes) -> void* {
        uintptr_t p = ((uintptr_t)wsp + 255) & ~(uintptr_t)255;
        wsp = (char*)(p + bytes);
        return (void*)p;
    };

    // ---- input cursor -------------------------------------------------
    int cur = 0;
    auto F = [&]() -> const float* { return (const float*)d_in[cur++]; };
    auto tohT = [&](const float* src, int K, int N, int* ldt_out) -> const _Float16* {
        const int ldt = (K + 7) & ~7;
        _Float16* dst = (_Float16*)alloc((size_t)N * ldt * sizeof(_Float16));
        int gx = (ldt + 255) / 256; if (gx > 64) gx = 64;
        dim3 grid(gx, N);
        b2v_w_to_f16_t<<<grid, 256, 0, stream>>>(src, dst, K, N, ldt);
        *ldt_out = ldt;
        return dst;
    };
    auto parseAttn = [&]() -> AttnW {
        AttnW a; int ldt;
        a.wq = tohT(F(), DMODEL, DMODEL, &ldt);
        a.wk = tohT(F(), DMODEL, DMODEL, &ldt);
        a.wv = tohT(F(), DMODEL, DMODEL, &ldt);
        a.wo = tohT(F(), DMODEL, DMODEL, &ldt);
        a.bq = F(); a.bk = F(); a.bv = F(); a.bo = F();
        return a;
    };
    auto parseLin = [&](int i, int o) -> LinW {
        LinW l; l.w = tohT(F(), i, o, &l.ldt); l.b = F(); return l;
    };
    auto parseLN = [&]() -> LNW { LNW l; l.g = F(); l.b = F(); return l; };

    // ---- unpack params (insertion order) ------------------------------
    const float* player_emb = F();
    const float* ball_emb   = F();

    LinW enc_pl0 = parseLin(MLP_IN, MLP_HID);
    LinW enc_pl1 = parseLin(MLP_HID, DMODEL);
    LinW enc_bl0 = parseLin(MLP_IN, MLP_HID);
    LinW enc_bl1 = parseLin(MLP_HID, DMODEL);
    EncL enc[NLAYERS];
    for (int l = 0; l < NLAYERS; ++l) {
        enc[l].sa  = parseAttn();
        enc[l].ln1 = parseLN();
        enc[l].ff1 = parseLin(DMODEL, DFF);
        enc[l].ff2 = parseLin(DFF, DMODEL);
        enc[l].ln2 = parseLN();
    }

    LinW dec_pl0 = parseLin(MLP_IN, MLP_HID);
    LinW dec_pl1 = parseLin(MLP_HID, DMODEL);
    LinW dec_bl0 = parseLin(MLP_IN, MLP_HID);
    LinW dec_bl1 = parseLin(MLP_HID, DMODEL);
    DecL dec[NLAYERS];
    for (int l = 0; l < NLAYERS; ++l) {
        dec[l].sa  = parseAttn();
        dec[l].ln1 = parseLN();
        dec[l].ca  = parseAttn();
        dec[l].ln2 = parseLN();
        dec[l].ff1 = parseLin(DMODEL, DFF);
        dec[l].ff2 = parseLin(DFF, DMODEL);
        dec[l].ln3 = parseLN();
    }

    LinW cls = parseLin(DMODEL, NLAB);

    const int*   player_idxs = (const int*)d_in[cur++];
    const float* player_xs   = F();
    const float* player_ys   = F();
    const float* player_hoop = F();
    const float* ball_xs     = F();
    const float* ball_ys     = F();
    const float* ball_zs     = F();

    // ---- activation buffers -------------------------------------------
    _Float16* feat16 = (_Float16*)alloc((size_t)NTOK * FEAT_LD * 2);
    _Float16* H16    = (_Float16*)alloc((size_t)NTOK * DFF * 2);
    float*    x      = (float*)   alloc((size_t)NTOK * DMODEL * 4);
    _Float16* xh     = (_Float16*)alloc((size_t)NTOK * DMODEL * 2);
    float*    delta  = (float*)   alloc((size_t)NTOK * DMODEL * 4);
    _Float16* memh   = (_Float16*)alloc((size_t)NTOK * DMODEL * 2);
    _Float16* Q      = (_Float16*)alloc((size_t)NTOK * DMODEL * 2);
    _Float16* Kb     = (_Float16*)alloc((size_t)NTOK * DMODEL * 2);
    _Float16* VT     = (_Float16*)alloc((size_t)NTOK * DMODEL * 2); // [DMODEL][NTOK]
    _Float16* O      = (_Float16*)alloc((size_t)NTOK * DMODEL * 2);
    float*    Sbuf   = (float*)   alloc((size_t)NTOK * NTOK * 4);
    _Float16* Pbuf   = (_Float16*)alloc((size_t)NTOK * NTOK * 2);

    // ================= encoder =================
    b2v_feats<<<NTOK, 64, 0, stream>>>(player_emb, player_idxs, player_xs,
                                       player_ys, player_hoop, ball_emb,
                                       ball_xs, ball_ys, ball_zs, 0, feat16);
    gemm(stream, feat16, FEAT_LD, FEAT_LD, enc_pl0.w, enc_pl0.ldt, enc_pl0.ldt,
         nullptr, 0, H16, MLP_HID, 0, enc_pl0.b, NPL_TOK, MLP_HID, MLP_IN, 1, 1.0f);
    gemm(stream, feat16 + (size_t)NPL_TOK * FEAT_LD, FEAT_LD, FEAT_LD,
         enc_bl0.w, enc_bl0.ldt, enc_bl0.ldt, nullptr, 0,
         H16 + (size_t)NPL_TOK * MLP_HID, MLP_HID, 0, enc_bl0.b,
         SEQ_LEN, MLP_HID, MLP_IN, 1, 1.0f);
    gemm(stream, H16, MLP_HID, MLP_HID, enc_pl1.w, enc_pl1.ldt, enc_pl1.ldt,
         x, DMODEL, xh, DMODEL, 0, enc_pl1.b, NPL_TOK, DMODEL, MLP_HID, 0, SQRT_D);
    gemm(stream, H16 + (size_t)NPL_TOK * MLP_HID, MLP_HID, MLP_HID,
         enc_bl1.w, enc_bl1.ldt, enc_bl1.ldt,
         x + (size_t)NPL_TOK * DMODEL, DMODEL,
         xh + (size_t)NPL_TOK * DMODEL, DMODEL, 0, enc_bl1.b,
         SEQ_LEN, DMODEL, MLP_HID, 0, SQRT_D);

    for (int l = 0; l < NLAYERS; ++l) {
        run_mha(stream, xh, xh, NTOK, NTOK, enc[l].sa, Sbuf, Pbuf,
                Q, Kb, VT, O, delta, /*masked=*/0);
        b2v_add_ln<<<NTOK, 256, 0, stream>>>(x, delta, enc[l].ln1.g, enc[l].ln1.b, x, xh);
        gemm(stream, xh, DMODEL, DMODEL, enc[l].ff1.w, enc[l].ff1.ldt, enc[l].ff1.ldt,
             nullptr, 0, H16, DFF, 0, enc[l].ff1.b, NTOK, DFF, DMODEL, 1, 1.0f);
        gemm(stream, H16, DFF, DFF, enc[l].ff2.w, enc[l].ff2.ldt, enc[l].ff2.ldt,
             delta, DMODEL, nullptr, 0, 0, enc[l].ff2.b, NTOK, DMODEL, DFF, 0, 1.0f);
        _Float16* o16 = (l == NLAYERS - 1) ? memh : xh;
        b2v_add_ln<<<NTOK, 256, 0, stream>>>(x, delta, enc[l].ln2.g, enc[l].ln2.b, x, o16);
    }

    // ================= decoder =================
    b2v_feats<<<NTOK, 64, 0, stream>>>(player_emb, player_idxs, player_xs,
                                       player_ys, player_hoop, ball_emb,
                                       ball_xs, ball_ys, ball_zs, 5, feat16);
    gemm(stream, feat16, FEAT_LD, FEAT_LD, dec_pl0.w, dec_pl0.ldt, dec_pl0.ldt,
         nullptr, 0, H16, MLP_HID, 0, dec_pl0.b, NPL_TOK, MLP_HID, MLP_IN, 1, 1.0f);
    gemm(stream, feat16 + (size_t)NPL_TOK * FEAT_LD, FEAT_LD, FEAT_LD,
         dec_bl0.w, dec_bl0.ldt, dec_bl0.ldt, nullptr, 0,
         H16 + (size_t)NPL_TOK * MLP_HID, MLP_HID, 0, dec_bl0.b,
         SEQ_LEN, MLP_HID, MLP_IN, 1, 1.0f);
    gemm(stream, H16, MLP_HID, MLP_HID, dec_pl1.w, dec_pl1.ldt, dec_pl1.ldt,
         x, DMODEL, xh, DMODEL, 0, dec_pl1.b, NPL_TOK, DMODEL, MLP_HID, 0, SQRT_D);
    gemm(stream, H16 + (size_t)NPL_TOK * MLP_HID, MLP_HID, MLP_HID,
         dec_bl1.w, dec_bl1.ldt, dec_bl1.ldt,
         x + (size_t)NPL_TOK * DMODEL, DMODEL,
         xh + (size_t)NPL_TOK * DMODEL, DMODEL, 0, dec_bl1.b,
         SEQ_LEN, DMODEL, MLP_HID, 0, SQRT_D);

    for (int l = 0; l < NLAYERS; ++l) {
        run_mha(stream, xh, xh, NTOK, NTOK, dec[l].sa, Sbuf, Pbuf,
                Q, Kb, VT, O, delta, /*masked=*/1);
        b2v_add_ln<<<NTOK, 256, 0, stream>>>(x, delta, dec[l].ln1.g, dec[l].ln1.b, x, xh);
        run_mha(stream, xh, memh, NTOK, NTOK, dec[l].ca, Sbuf, Pbuf,
                Q, Kb, VT, O, delta, /*masked=*/0);
        b2v_add_ln<<<NTOK, 256, 0, stream>>>(x, delta, dec[l].ln2.g, dec[l].ln2.b, x, xh);
        gemm(stream, xh, DMODEL, DMODEL, dec[l].ff1.w, dec[l].ff1.ldt, dec[l].ff1.ldt,
             nullptr, 0, H16, DFF, 0, dec[l].ff1.b, NTOK, DFF, DMODEL, 1, 1.0f);
        gemm(stream, H16, DFF, DFF, dec[l].ff2.w, dec[l].ff2.ldt, dec[l].ff2.ldt,
             delta, DMODEL, nullptr, 0, 0, dec[l].ff2.b, NTOK, DMODEL, DFF, 0, 1.0f);
        b2v_add_ln<<<NTOK, 256, 0, stream>>>(x, delta, dec[l].ln3.g, dec[l].ln3.b, x, xh);
    }

    // ================= classifier =================
    gemm(stream, xh, DMODEL, DMODEL, cls.w, cls.ldt, cls.ldt,
         (float*)d_out, NLAB, nullptr, 0, 0, cls.b, NTOK, NLAB, DMODEL, 0, 1.0f);
}
